// NonLocalBlock_87299505258783
// MI455X (gfx1250) — compile-verified, matrix operand order checked by reading, not run on hardware
//
#include <hip/hip_runtime.h>
#include <hip/hip_bf16.h>

// NonLocalBlock on MI455X (gfx1250, wave32, WMMA bf16 + TDM staging).
// Needs ws_size >= ~146 MB (152,047,616 bytes).

typedef __bf16 bf16_t;
typedef __attribute__((ext_vector_type(16))) __bf16 v16bf;
typedef __attribute__((ext_vector_type(8)))  __bf16 v8bf;
typedef __attribute__((ext_vector_type(8)))  float  v8f;

#if __has_builtin(__builtin_amdgcn_tensor_load_to_lds) && \
    __has_builtin(__builtin_amdgcn_s_wait_tensorcnt)
#define HAVE_TDM 1
#endif

__device__ __forceinline__ v8f vzero8() {
  v8f z;
#pragma unroll
  for (int i = 0; i < 8; ++i) z[i] = 0.f;
  return z;
}

// Fragment load for 16-bit 16x32 WMMA operands from a row-major 32-wide row:
// lane half h -> two contiguous 16B runs at byte offsets 16h and 32+16h.
__device__ __forceinline__ v16bf frag_ld(const bf16_t* rowBase, int h) {
  v8bf lo = *(const v8bf*)(rowBase + 8 * h);
  v8bf hi = *(const v8bf*)(rowBase + 16 + 8 * h);
  v16bf r;
#pragma unroll
  for (int i = 0; i < 8; ++i) { r[i] = lo[i]; r[i + 8] = hi[i]; }
  return r;
}

__device__ __forceinline__ v8f wmma_bf16(v16bf a, v16bf b, v8f c) {
  return __builtin_amdgcn_wmma_f32_16x16x32_bf16(false, a, false, b, (short)0, c,
                                                 false, false);
}

__device__ __forceinline__ unsigned pack2(float a, float b) {
  unsigned short ul = __builtin_bit_cast(unsigned short, (bf16_t)a);
  unsigned short uh = __builtin_bit_cast(unsigned short, (bf16_t)b);
  return (unsigned)ul | ((unsigned)uh << 16);  // lowers to v_cvt_pk_bf16_f32
}

#ifdef HAVE_TDM
typedef unsigned int v4u __attribute__((ext_vector_type(4)));
typedef int v4i __attribute__((ext_vector_type(4)));
typedef int v8i __attribute__((ext_vector_type(8)));

// 2D bf16 tile load: tile_x elems per row, 'rows' rows, row stride in elems.
// D# per CDNA5 ISA ch.8: group0 {count=1, lds_addr, global_addr, type=2},
// group1 {data_size=2B, tensor dims == tile dims, dim0 stride}.
__device__ __forceinline__ void tdm_load_2d(unsigned lds_off, const void* gptr,
                                            unsigned tile_x, unsigned rows,
                                            unsigned stride) {
  unsigned long long ga = (unsigned long long)(uintptr_t)gptr;
  v4u g0;
  g0[0] = 1u;                                     // count=1, user mode
  g0[1] = lds_off;                                // lds_addr (bytes)
  g0[2] = (unsigned)(ga & 0xFFFFFFFFu);           // global_addr[31:0]
  g0[3] = (unsigned)((ga >> 32) & 0x01FFFFFFu) | 0x80000000u;  // addr[56:32]|type=2
  v8i g1;
  g1[0] = (int)0x00010000u;                       // data_size=1 (2 bytes)
  g1[1] = (int)((tile_x & 0xFFFFu) << 16);        // tensor_dim0 lo16
  g1[2] = (int)(((tile_x >> 16) & 0xFFFFu) | ((rows & 0xFFFFu) << 16));
  g1[3] = (int)(((rows >> 16) & 0xFFFFu) | (tile_x << 16));    // dim1 hi | tile_dim0
  g1[4] = (int)(rows & 0xFFFFu);                  // tile_dim1 (tile_dim2=0)
  g1[5] = (int)stride;                            // tensor_dim0_stride lo32
  g1[6] = 0;
  g1[7] = 0;
  v4i z4 = {0, 0, 0, 0};
#if __clang_major__ >= 23
  v8i z8 = {0, 0, 0, 0, 0, 0, 0, 0};
  __builtin_amdgcn_tensor_load_to_lds(g0, g1, z4, z4, z8, 0);
#else
  __builtin_amdgcn_tensor_load_to_lds(g0, g1, z4, z4, 0);
#endif
}
#endif  // HAVE_TDM

// ---------------------------------------------------------------- K0: prep
__global__ void k0_prep(const float* __restrict__ g_w, const float* __restrict__ theta_w,
                        const float* __restrict__ phi_w, const float* __restrict__ w_w,
                        bf16_t* __restrict__ wG, bf16_t* __restrict__ wW,
                        float* __restrict__ stats) {
  int idx = blockIdx.x * 256 + threadIdx.x;
  if (idx < 768 * 512) {
    int o = idx >> 9, c = idx & 511;
    float v = (o < 256) ? g_w[o * 512 + c]
            : (o < 512) ? theta_w[(o - 256) * 512 + c]
                        : phi_w[(o - 512) * 512 + c];
    wG[idx] = (bf16_t)v;
  } else if (idx < 768 * 512 + 512 * 256) {
    int i = idx - 768 * 512;
    wW[i] = (bf16_t)w_w[i];
  } else if (idx < 768 * 512 + 512 * 256 + 1024) {
    stats[idx - (768 * 512 + 512 * 256)] = 0.f;  // re-zeroed every call
  }
}

// -------------------------------------------- K1: fused g/theta/phi projection
// grid (32 n-tiles of 128, 12 o-groups of 64, 16 batch), block 256 (8 waves)
__global__ __launch_bounds__(256) void k1_proj(
    const float* __restrict__ x, const bf16_t* __restrict__ wG,
    const float* __restrict__ g_b, const float* __restrict__ theta_b,
    const float* __restrict__ phi_b, bf16_t* __restrict__ thetaO,
    bf16_t* __restrict__ gO, bf16_t* __restrict__ phiO) {
  __shared__ __align__(16) bf16_t Xs[128 * 32];   // [n][k]
  __shared__ __align__(16) bf16_t Ws[64 * 32];    // [o][k]
  __shared__ float Outs[64 * 128];                // [o][n]
  const int t = threadIdx.x;
  const int nt = blockIdx.x, og = blockIdx.y, b = blockIdx.z;
  const int n0 = nt * 128;
  const int w = t >> 5, lane = t & 31, col = lane & 15, hv = lane >> 4;
  const int ot = w >> 1;
  const int nbase = (w & 1) * 4;

  v8f acc[4];
#pragma unroll
  for (int j = 0; j < 4; ++j) acc[j] = vzero8();

  for (int kc = 0; kc < 16; ++kc) {
    __syncthreads();
    // X tile: fp32 -> packed bf16 pairs (v_cvt_pk_bf16_f32 + one b32 DS store)
#pragma unroll
    for (int i = 0; i < 8; ++i) {
      int linear = t + 256 * i;
      int n = linear & 127, kp = linear >> 7;  // kp 0..15
      const float* xp = &x[(((size_t)b * 512 + kc * 32 + 2 * kp) * 4096) + n0 + n];
      *(unsigned*)&Xs[n * 32 + 2 * kp] = pack2(xp[0], xp[4096]);
    }
#ifdef HAVE_TDM
    if (t < 32) {  // W tile via Tensor Data Mover: 64 rows x 64B, stride 1KB
      tdm_load_2d((unsigned)(uintptr_t)Ws, wG + (size_t)(og * 64) * 512 + kc * 32,
                  32, 64, 512);
      __builtin_amdgcn_s_wait_tensorcnt(0);
    }
#else
    {
      int r = t >> 2, ch = t & 3;
      *(uint4*)&Ws[r * 32 + ch * 8] =
          *(const uint4*)&wG[(size_t)(og * 64 + r) * 512 + kc * 32 + ch * 8];
    }
#endif
    if (kc < 15) {  // near-scope prefetch of next x chunk (global_prefetch_b8)
      int n = t & 127, kp = t >> 7;
      __builtin_prefetch(
          &x[(((size_t)b * 512 + (kc + 1) * 32 + 2 * kp) * 4096) + n0 + n], 0, 3);
    }
    __syncthreads();
    v16bf a = frag_ld(&Ws[(ot * 16 + col) * 32], hv);
    v16bf bbr[4];
#pragma unroll
    for (int j = 0; j < 4; ++j)
      bbr[j] = frag_ld(&Xs[((nbase + j) * 16 + col) * 32], hv);
#pragma unroll
    for (int j = 0; j < 4; ++j) acc[j] = wmma_bf16(a, bbr[j], acc[j]);
  }
#pragma unroll
  for (int j = 0; j < 4; ++j)
#pragma unroll
    for (int v = 0; v < 8; ++v)
      Outs[(ot * 16 + v + 8 * hv) * 128 + (nbase + j) * 16 + col] = acc[j][v];
  __syncthreads();

  if (og < 4) {         // g: bias + 2x2 maxpool -> [b][ci][m]
    int cib = og * 64;
#pragma unroll
    for (int i = 0; i < 8; ++i) {
      int linear = t + 256 * i;
      int o = linear >> 5, wp = linear & 31;
      float m0 = Outs[o * 128 + 2 * wp],      m1 = Outs[o * 128 + 2 * wp + 1];
      float m2 = Outs[o * 128 + 64 + 2 * wp], m3 = Outs[o * 128 + 64 + 2 * wp + 1];
      float mv = fmaxf(fmaxf(m0, m1), fmaxf(m2, m3)) + g_b[cib + o];
      gO[((size_t)(b * 256 + cib + o)) * 1024 + nt * 32 + wp] = (bf16_t)mv;
    }
  } else if (og < 8) {  // theta -> [b][n][ci]
    int cib = (og - 4) * 64;
    for (int i = 0; i < 32; ++i) {
      int linear = t + 256 * i;
      int o = linear & 63, n = linear >> 6;
      float v = Outs[o * 128 + n] + theta_b[cib + o];
      thetaO[((size_t)(b * 4096 + n0 + n)) * 256 + cib + o] = (bf16_t)v;
    }
  } else {              // phi: bias + pool -> [b][m][ci]
    int cib = (og - 8) * 64;
#pragma unroll
    for (int i = 0; i < 8; ++i) {
      int linear = t + 256 * i;
      int o = linear & 63, wp = linear >> 6;
      float m0 = Outs[o * 128 + 2 * wp],      m1 = Outs[o * 128 + 2 * wp + 1];
      float m2 = Outs[o * 128 + 64 + 2 * wp], m3 = Outs[o * 128 + 64 + 2 * wp + 1];
      float mv = fmaxf(fmaxf(m0, m1), fmaxf(m2, m3)) + phi_b[cib + o];
      phiO[((size_t)(b * 1024 + nt * 32 + wp)) * 256 + cib + o] = (bf16_t)mv;
    }
  }
}

// ---------------------------------------------- K2: flash attention (R=64, Mc=32)
__global__ __launch_bounds__(256) void k2_attn(
    const bf16_t* __restrict__ thetaI, const bf16_t* __restrict__ gI,
    const bf16_t* __restrict__ phiI, bf16_t* __restrict__ yO) {
  __shared__ __align__(16) bf16_t Th[64 * 256];   // theta rows [r][k]
  __shared__ __align__(16) bf16_t PG[32 * 256];   // phi chunk [m][k] / gT [ci][m]
  __shared__ float Sc[64 * 32];
  __shared__ __align__(16) bf16_t Pc[64 * 32];
  __shared__ float rowm[64], rowl[64], rowscale[64];
  __shared__ float pred[64 * 4];
  const int t = threadIdx.x;
  const int ntile = blockIdx.x, b = blockIdx.y;
  const int n0 = ntile * 64;
  const int w = t >> 5, lane = t & 31, col = lane & 15, hv = lane >> 4;
  const int rt = w >> 1;
  const int cib = (w & 1) * 128;

#ifdef HAVE_TDM
  if (t < 32) {  // theta block: one contiguous 32KB TDM transfer
    tdm_load_2d((unsigned)(uintptr_t)Th, thetaI + (size_t)(b * 4096 + n0) * 256,
                16384, 1, 16384);
    __builtin_amdgcn_s_wait_tensorcnt(0);
  }
#else
  for (int i = 0; i < 8; ++i) {
    int idx = t + 256 * i;
    ((uint4*)Th)[idx] = ((const uint4*)(thetaI + (size_t)(b * 4096 + n0) * 256))[idx];
  }
#endif
  if (t < 64) { rowm[t] = -3.0e38f; rowl[t] = 0.f; }
  v8f acc[8];
#pragma unroll
  for (int j = 0; j < 8; ++j) acc[j] = vzero8();
  __syncthreads();

  for (int mc = 0; mc < 32; ++mc) {
#ifdef HAVE_TDM
    if (t < 32) {  // phi chunk: contiguous 16KB
      tdm_load_2d((unsigned)(uintptr_t)PG, phiI + (size_t)(b * 1024 + mc * 32) * 256,
                  8192, 1, 8192);
      __builtin_amdgcn_s_wait_tensorcnt(0);
    }
#else
    for (int i = 0; i < 4; ++i) {
      int idx = t + 256 * i;
      ((uint4*)PG)[idx] =
          ((const uint4*)(phiI + (size_t)(b * 1024 + mc * 32) * 256))[idx];
    }
#endif
    __syncthreads();
    {  // S(rt, mt) = Th * Phi^T, K=256, two independent WMMA chains
      const int mt = w & 1;
      v8f s0 = vzero8(), s1 = vzero8();
#pragma unroll
      for (int kc = 0; kc < 8; kc += 2) {
        v16bf a0 = frag_ld(&Th[(rt * 16 + col) * 256 + kc * 32], hv);
        v16bf b0 = frag_ld(&PG[(mt * 16 + col) * 256 + kc * 32], hv);
        v16bf a1 = frag_ld(&Th[(rt * 16 + col) * 256 + (kc + 1) * 32], hv);
        v16bf b1 = frag_ld(&PG[(mt * 16 + col) * 256 + (kc + 1) * 32], hv);
        s0 = wmma_bf16(a0, b0, s0);
        s1 = wmma_bf16(a1, b1, s1);
      }
      v8f s = s0 + s1;
#pragma unroll
      for (int v = 0; v < 8; ++v)
        Sc[(rt * 16 + v + 8 * hv) * 32 + mt * 16 + col] = s[v];
    }
    __syncthreads();
#ifdef HAVE_TDM
    if (t < 32) {  // gT chunk: 256 rows x 64B, stride 2KB (2D TDM)
      tdm_load_2d((unsigned)(uintptr_t)PG, gI + (size_t)b * 256 * 1024 + mc * 32,
                  32, 256, 1024);
      __builtin_amdgcn_s_wait_tensorcnt(0);
    }
#else
    for (int i = 0; i < 4; ++i) {
      int idx = t + 256 * i;
      int ci = idx >> 2, ch = idx & 3;
      *(uint4*)&PG[ci * 32 + ch * 8] =
          *(const uint4*)&gI[(size_t)(b * 256 + ci) * 1024 + mc * 32 + ch * 8];
    }
#endif
    {  // softmax partial max: 4 threads/row
      int r = t >> 2, q = t & 3;
      float pm = -3.0e38f;
#pragma unroll
      for (int e = 0; e < 8; ++e) pm = fmaxf(pm, Sc[r * 32 + q * 8 + e]);
      pred[r * 4 + q] = pm;
    }
    __syncthreads();
    if (t < 64) {
      float mn = fmaxf(fmaxf(pred[t * 4], pred[t * 4 + 1]),
                       fmaxf(pred[t * 4 + 2], pred[t * 4 + 3]));
      mn = fmaxf(mn, rowm[t]);
      rowscale[t] = __expf(rowm[t] - mn);
      rowm[t] = mn;
    }
    __syncthreads();
    {  // exp + partial sums, write P chunk
      int r = t >> 2, q = t & 3;
      float mr = rowm[r], s = 0.f;
#pragma unroll
      for (int e = 0; e < 8; ++e) {
        float ev = __expf(Sc[r * 32 + q * 8 + e] - mr);
        Pc[r * 32 + q * 8 + e] = (bf16_t)ev;
        s += ev;
      }
      pred[r * 4 + q] = s;
    }
    __syncthreads();
    if (t < 64)
      rowl[t] = rowl[t] * rowscale[t] +
                (pred[t * 4] + pred[t * 4 + 1] + pred[t * 4 + 2] + pred[t * 4 + 3]);
    __syncthreads();
    // rescale running Y, then Y += P * G (K=32, one WMMA step per tile)
#pragma unroll
    for (int v = 0; v < 8; ++v) {
      float scv = rowscale[rt * 16 + v + 8 * hv];
#pragma unroll
      for (int j = 0; j < 8; ++j) acc[j][v] *= scv;
    }
    {
      v16bf a = frag_ld(&Pc[(rt * 16 + col) * 32], hv);
      v16bf bb[4];
#pragma unroll
      for (int j = 0; j < 4; ++j)
        bb[j] = frag_ld(&PG[((cib + j * 16) + col) * 32], hv);
#pragma unroll
      for (int j = 0; j < 4; ++j) acc[j] = wmma_bf16(a, bb[j], acc[j]);
#pragma unroll
      for (int j = 0; j < 4; ++j)
        bb[j] = frag_ld(&PG[((cib + (j + 4) * 16) + col) * 32], hv);
#pragma unroll
      for (int j = 0; j < 4; ++j) acc[j + 4] = wmma_bf16(a, bb[j], acc[j + 4]);
    }
    __syncthreads();
  }
#pragma unroll
  for (int v = 0; v < 8; ++v) {
    int r = rt * 16 + v + 8 * hv;
    float inv = 1.0f / rowl[r];
#pragma unroll
    for (int j = 0; j < 8; ++j)
      yO[((size_t)(b * 4096 + n0 + r)) * 256 + cib + j * 16 + col] =
          (bf16_t)(acc[j][v] * inv);
  }
}

// ----------------------------------------------- K3: W projection + BN stats
__global__ __launch_bounds__(256) void k3_wproj(
    const bf16_t* __restrict__ yI, const bf16_t* __restrict__ wW,
    const float* __restrict__ w_b, bf16_t* __restrict__ wyO,
    float* __restrict__ stats) {
  __shared__ __align__(16) bf16_t Ys[128 * 32];  // [n][k]
  __shared__ __align__(16) bf16_t Ws2[64 * 32];  // [c][k]
  __shared__ float Outs[128 * 64];               // [n][c]
  const int t = threadIdx.x;
  const int nt = blockIdx.x, cg = blockIdx.y, b = blockIdx.z;
  const int n0 = nt * 128;
  const int w = t >> 5, lane = t & 31, col = lane & 15, hv = lane >> 4;
  v8f acc[4];
#pragma unroll
  for (int j = 0; j < 4; ++j) acc[j] = vzero8();

  for (int kc = 0; kc < 8; ++kc) {
    __syncthreads();
#ifdef HAVE_TDM
    if (t < 32) {  // y tile: 128 rows x 64B (stride 512B); W tile: 64 rows
      tdm_load_2d((unsigned)(uintptr_t)Ys,
                  yI + (size_t)(b * 4096 + n0) * 256 + kc * 32, 32, 128, 256);
      tdm_load_2d((unsigned)(uintptr_t)Ws2,
                  wW + (size_t)(cg * 64) * 256 + kc * 32, 32, 64, 256);
      __builtin_amdgcn_s_wait_tensorcnt(0);
    }
#else
#pragma unroll
    for (int i = 0; i < 2; ++i) {
      int idx = t + 256 * i;
      int n = idx >> 2, ch = idx & 3;
      *(uint4*)&Ys[n * 32 + ch * 8] =
          *(const uint4*)&yI[(size_t)(b * 4096 + n0 + n) * 256 + kc * 32 + ch * 8];
    }
    {
      int c = t >> 2, ch = t & 3;
      *(uint4*)&Ws2[c * 32 + ch * 8] =
          *(const uint4*)&wW[(size_t)(cg * 64 + c) * 256 + kc * 32 + ch * 8];
    }
#endif
    if (kc < 7) {  // near-scope prefetch of next y chunk
      int n = t >> 1, half = t & 1;
      __builtin_prefetch(
          &yI[(size_t)(b * 4096 + n0 + n) * 256 + (kc + 1) * 32 + half * 16], 0, 3);
    }
    __syncthreads();
    v16bf a = frag_ld(&Ys[(w * 16 + col) * 32], hv);
    v16bf bbr[4];
#pragma unroll
    for (int j = 0; j < 4; ++j) bbr[j] = frag_ld(&Ws2[(j * 16 + col) * 32], hv);
#pragma unroll
    for (int j = 0; j < 4; ++j) acc[j] = wmma_bf16(a, bbr[j], acc[j]);
  }
  __syncthreads();
#pragma unroll
  for (int j = 0; j < 4; ++j)
#pragma unroll
    for (int v = 0; v < 8; ++v)
      Outs[(w * 16 + v + 8 * hv) * 64 + j * 16 + col] = acc[j][v];
  __syncthreads();
  for (int i = 0; i < 32; ++i) {  // store w_y [b][c][n], coalesced in n
    int linear = t + 256 * i;
    int n = linear & 127, c = linear >> 7;
    float v = Outs[n * 64 + c] + w_b[cg * 64 + c];
    wyO[((size_t)(b * 512 + cg * 64 + c)) * 4096 + n0 + n] = (bf16_t)v;
  }
  if (t < 64) {  // per-channel partial sums for BN (native f32 atomics)
    float bias = w_b[cg * 64 + t];
    float s = 0.f, sq = 0.f;
    for (int n = 0; n < 128; ++n) {
      float v = Outs[n * 64 + t] + bias;
      s += v; sq += v * v;
    }
    unsafeAtomicAdd(&stats[cg * 64 + t], s);
    unsafeAtomicAdd(&stats[512 + cg * 64 + t], sq);
  }
}

// ----------------------------------------------------- K4: BN + residual
__global__ __launch_bounds__(256) void k4_bn(
    const bf16_t* __restrict__ wy, const float* __restrict__ x,
    const float* __restrict__ stats, const float* __restrict__ gamma,
    const float* __restrict__ beta, float* __restrict__ out) {
  size_t base = ((size_t)blockIdx.x * 256 + threadIdx.x) * 8;
  int c = (int)((base >> 12) & 511);
  float mean = stats[c] * (1.0f / 65536.0f);
  float var = stats[512 + c] * (1.0f / 65536.0f) - mean * mean;
  float inv = rsqrtf(var + 1e-5f) * gamma[c];
  float bt = beta[c];
  v8bf wv = *(const v8bf*)(wy + base);
  float xs[8], os[8];
  *(float4*)&xs[0] = *(const float4*)(x + base);
  *(float4*)&xs[4] = *(const float4*)(x + base + 4);
#pragma unroll
  for (int i = 0; i < 8; ++i)
    os[i] = ((float)wv[i] - mean) * inv + bt + xs[i];
  *(float4*)(out + base) = *(float4*)&os[0];
  *(float4*)(out + base + 4) = *(float4*)&os[4];
}

extern "C" void kernel_launch(void* const* d_in, const int* in_sizes, int n_in,
                              void* d_out, int out_size, void* d_ws, size_t ws_size,
                              hipStream_t stream) {
  const float* x       = (const float*)d_in[0];
  const float* g_w     = (const float*)d_in[1];
  const float* g_b     = (const float*)d_in[2];
  const float* theta_w = (const float*)d_in[3];
  const float* theta_b = (const float*)d_in[4];
  const float* phi_w   = (const float*)d_in[5];
  const float* phi_b   = (const float*)d_in[6];
  const float* w_w     = (const float*)d_in[7];
  const float* w_b     = (const float*)d_in[8];
  const float* bn_g    = (const float*)d_in[9];
  const float* bn_b    = (const float*)d_in[10];

  char* ws = (char*)d_ws;
  bf16_t* wG     = (bf16_t*)(ws + 0);           // 768x512
  bf16_t* wWb    = (bf16_t*)(ws + 786432);      // 512x256
  bf16_t* thetaB = (bf16_t*)(ws + 1048576);     // [B][N][CI]
  bf16_t* gB     = (bf16_t*)(ws + 34603008);    // [B][CI][M]
  bf16_t* phiB   = (bf16_t*)(ws + 42991616);    // [B][M][CI]
  bf16_t* yB     = (bf16_t*)(ws + 51380224);    // [B][N][CI]
  bf16_t* wyB    = (bf16_t*)(ws + 84934656);    // [B][C][N]
  float*  stats  = (float*)(ws + 152043520);    // 2x512
  float* out = (float*)d_out;

  k0_prep<<<2052, 256, 0, stream>>>(g_w, theta_w, phi_w, w_w, wG, wWb, stats);
  k1_proj<<<dim3(32, 12, 16), 256, 0, stream>>>(x, wG, g_b, theta_b, phi_b,
                                                thetaB, gB, phiB);
  k2_attn<<<dim3(64, 16), 256, 0, stream>>>(thetaB, gB, phiB, yB);
  k3_wproj<<<dim3(32, 8, 16), 256, 0, stream>>>(yB, wWb, w_b, wyB, stats);
  k4_bn<<<16384, 256, 0, stream>>>(wyB, x, stats, bn_g, bn_b, out);
}
